// KeyAttention_9311489098350
// MI455X (gfx1250) — compile-verified, matrix operand order checked by reading, not run on hardware
//
#include <hip/hip_runtime.h>
#include <hip/hip_bf16.h>

// ---------------------------------------------------------------------------
// KeyAttention for MI455X (gfx1250, wave32, WMMA bf16 with hi/lo fp32 split)
// B=128, TA=256, TK=256, E=1024
// ---------------------------------------------------------------------------

#define B_   128
#define TA_  256
#define TK_  256
#define E_   1024
#define NEGC (-10000.0f)

typedef __attribute__((ext_vector_type(16))) __bf16 v16bf;
typedef __attribute__((ext_vector_type(8)))  __bf16 v8bf;
typedef __attribute__((ext_vector_type(8)))  float  v8f;

static __device__ __forceinline__ v8f wmma_bf16(v16bf a, v16bf b, v8f c) {
    // (neg_a, A, neg_b, B, c_mod, C, reuse_a, reuse_b)
    return __builtin_amdgcn_wmma_f32_16x16x32_bf16(false, a, false, b,
                                                   (short)0, c, false, false);
}

// Prefetch next K-chunk into the cache hierarchy (lowers to global_prefetch_b8;
// no LOADcnt cost, overlaps with the current chunk's WMMAs).
static __device__ __forceinline__ void prefetch_g(const void* p) {
    __builtin_prefetch(p, 0, 3);
}

// Load one 16x32 bf16 fragment (A or column-major B) from an LDS tile stored
// row-major with row stride 32 bf16 (64 bytes). Per CDNA5 ISA layout:
// lane = (lane&15) selects the row, (lane>>4)*8 selects the K sub-block, and
// the 16 elements are two contiguous 16-byte chunks at +0B and +32B.
static __device__ __forceinline__ v16bf ldfrag(const __bf16* base, int row) {
    const int lane = threadIdx.x & 31;
    const int r    = row + (lane & 15);
    const int kb   = (lane >> 4) * 8;
    const v8bf* p0 = (const v8bf*)(base + r * 32 + kb);
    const v8bf* p1 = (const v8bf*)(base + r * 32 + kb + 16);
    v8bf x = *p0, y = *p1;
    return __builtin_shufflevector(x, y, 0, 1, 2, 3, 4, 5, 6, 7,
                                   8, 9, 10, 11, 12, 13, 14, 15);
}

static __device__ __forceinline__ void split_bf16(float x, __bf16& h, __bf16& l) {
    h = (__bf16)x;
    l = (__bf16)(x - (float)h);
}

// ---------------------------------------------------------------------------
// K0: masked row norms for Z_cos.  One wave per row of E=1024.
// ---------------------------------------------------------------------------
__global__ void __launch_bounds__(256)
norm_kernel(const float* __restrict__ ans, const float* __restrict__ mask_ans,
            const float* __restrict__ key, const float* __restrict__ mask_key,
            float* __restrict__ norm_ans, float* __restrict__ norm_key) {
    const int w = threadIdx.x >> 5, lane = threadIdx.x & 31;
    const int r = blockIdx.x * 8 + w;
    const float* src;
    float m;
    float* dst;
    if (r < B_ * TA_) {
        src = ans + (size_t)r * E_;
        m   = mask_ans[r];
        dst = norm_ans + r;
    } else {
        const int r2 = r - B_ * TA_;
        src = key + (size_t)r2 * E_;
        m   = mask_key[r2];
        dst = norm_key + r2;
    }
    float s = 0.f;
#pragma unroll 8
    for (int j = 0; j < E_ / 32; j++) {
        float x = src[lane + 32 * j];
        s += x * x;
    }
    for (int o = 16; o > 0; o >>= 1) s += __shfl_xor(s, o, 32);
    if (lane == 0) *dst = sqrtf(fmaxf(m * s, 1e-8f));
}

// ---------------------------------------------------------------------------
// K1: Z = (mask*ans) @ (mask*key)^T  per batch.  128x128 output tile per
// block, 8 waves each owning a 32x64 sub-tile (2x4 16x16 frags).  fp32 is
// emulated with a 3-term bf16 hi/lo split per 32-wide K chunk.
// ---------------------------------------------------------------------------
__global__ void __launch_bounds__(256)
z_gemm_kernel(const float* __restrict__ ans, const float* __restrict__ key,
              const float* __restrict__ mask_ans, const float* __restrict__ mask_key,
              float* __restrict__ Z) {
    __shared__ __align__(16) __bf16 Ah[128 * 32], Al[128 * 32];
    __shared__ __align__(16) __bf16 Bh[128 * 32], Bl[128 * 32];

    const int b   = blockIdx.y;
    const int ta0 = (blockIdx.x >> 1) * 128;
    const int tk0 = (blockIdx.x & 1) * 128;
    const int tid = threadIdx.x;
    const int w = tid >> 5, lane = tid & 31;
    const int wr = w >> 1, wc = w & 1;   // 4x2 wave grid -> 32x64 per wave

    v8f acc[2][4];
#pragma unroll
    for (int i = 0; i < 2; i++)
#pragma unroll
        for (int j = 0; j < 4; j++) acc[i][j] = {};

    for (int e0 = 0; e0 < E_; e0 += 32) {
        // Stage A (128x32) and B (128x32, B holds key rows == Z columns).
#pragma unroll
        for (int q = 0; q < 4; q++) {
            const int idx = tid + 256 * q;
            const int row = idx >> 3;
            const int kq  = (idx & 7) * 4;
            const float* pa = ans + ((size_t)(b * TA_ + ta0 + row)) * E_ + e0 + kq;
            const float* pk = key + ((size_t)(b * TK_ + tk0 + row)) * E_ + e0 + kq;
            {
                const float m = mask_ans[b * TA_ + ta0 + row];
                const float4 v = *(const float4*)pa;
                float xs[4] = {v.x * m, v.y * m, v.z * m, v.w * m};
#pragma unroll
                for (int i = 0; i < 4; i++)
                    split_bf16(xs[i], Ah[row * 32 + kq + i], Al[row * 32 + kq + i]);
            }
            {
                const float m = mask_key[b * TK_ + tk0 + row];
                const float4 v = *(const float4*)pk;
                float xs[4] = {v.x * m, v.y * m, v.z * m, v.w * m};
#pragma unroll
                for (int i = 0; i < 4; i++)
                    split_bf16(xs[i], Bh[row * 32 + kq + i], Bl[row * 32 + kq + i]);
            }
            if (e0 + 32 < E_) {           // pull next chunk toward the WGP
                prefetch_g(pa + 32);
                prefetch_g(pk + 32);
            }
        }
        __syncthreads();

        v16bf ah[2], al[2], bh[4], bl[4];
#pragma unroll
        for (int i = 0; i < 2; i++) {
            ah[i] = ldfrag(Ah, wr * 32 + 16 * i);
            al[i] = ldfrag(Al, wr * 32 + 16 * i);
        }
#pragma unroll
        for (int j = 0; j < 4; j++) {
            bh[j] = ldfrag(Bh, wc * 64 + 16 * j);
            bl[j] = ldfrag(Bl, wc * 64 + 16 * j);
        }
#pragma unroll
        for (int i = 0; i < 2; i++)
#pragma unroll
            for (int j = 0; j < 4; j++) {
                acc[i][j] = wmma_bf16(ah[i], bh[j], acc[i][j]);  // hi*hi
                acc[i][j] = wmma_bf16(ah[i], bl[j], acc[i][j]);  // hi*lo
                acc[i][j] = wmma_bf16(al[i], bh[j], acc[i][j]);  // lo*hi
            }
        __syncthreads();
    }

    // Epilogue: C/D layout -> lane&15 = N, (lane>>4)*8 + vgpr = M.
#pragma unroll
    for (int i = 0; i < 2; i++)
#pragma unroll
        for (int j = 0; j < 4; j++) {
            const int col     = tk0 + wc * 64 + 16 * j + (lane & 15);
            const int rowbase = ta0 + wr * 32 + 16 * i + ((lane >> 4) * 8);
#pragma unroll
            for (int r = 0; r < 8; r++)
                Z[((size_t)(b * TA_ + rowbase + r)) * TK_ + col] = acc[i][j][r];
        }
}

// ---------------------------------------------------------------------------
// K2: softmax over key axis (per (b,a) row) + beta_key; writes P_key^T.
// ---------------------------------------------------------------------------
__global__ void __launch_bounds__(256)
softmax_key_kernel(const float* __restrict__ Z, const float* __restrict__ mask_key,
                   const float* __restrict__ na, const float* __restrict__ nk,
                   float* __restrict__ PkeyT, float* __restrict__ beta_key) {
    const int w = threadIdx.x >> 5, lane = threadIdx.x & 31;
    const int rid = blockIdx.x * 8 + w;
    const int b = rid >> 8, a = rid & 255;
    const float* zrow = Z + ((size_t)(b * TA_ + a)) * TK_;
    const float invna = 1.f / na[b * TA_ + a];

    float z[8], minf[8];
    float m = -3.0e38f, cmax = -3.0e38f;
#pragma unroll
    for (int j = 0; j < 8; j++) {
        const int k = lane + 32 * j;
        z[j]    = zrow[k];
        minf[j] = (1.f - mask_key[b * TK_ + k]) * NEGC;
        m       = fmaxf(m, z[j] + minf[j]);
        cmax    = fmaxf(cmax, z[j] * invna / nk[b * TK_ + k] + minf[j]);
    }
    for (int o = 16; o > 0; o >>= 1) {
        m    = fmaxf(m, __shfl_xor(m, o, 32));
        cmax = fmaxf(cmax, __shfl_xor(cmax, o, 32));
    }
    float e[8], s = 0.f;
#pragma unroll
    for (int j = 0; j < 8; j++) {
        e[j] = __expf(z[j] + minf[j] - m);
        s += e[j];
    }
    for (int o = 16; o > 0; o >>= 1) s += __shfl_xor(s, o, 32);
    const float inv = 1.f / s;
#pragma unroll
    for (int j = 0; j < 8; j++) {
        const int k = lane + 32 * j;
        PkeyT[((size_t)(b * TK_ + k)) * TA_ + a] = e[j] * inv;
    }
    if (lane == 0) beta_key[b * TA_ + a] = 1.f / (1.f + __expf(-5.f * cmax));
}

// ---------------------------------------------------------------------------
// K3: softmax over ans axis (per (b,k) column) + beta_ans; writes P_ans.
// ---------------------------------------------------------------------------
__global__ void __launch_bounds__(256)
softmax_ans_kernel(const float* __restrict__ Z, const float* __restrict__ mask_ans,
                   const float* __restrict__ na, const float* __restrict__ nk,
                   float* __restrict__ Pans, float* __restrict__ beta_ans) {
    const int w = threadIdx.x >> 5, lane = threadIdx.x & 31;
    const int rid = blockIdx.x * 8 + w;
    const int b = rid >> 8, k = rid & 255;
    const float invnk = 1.f / nk[b * TK_ + k];

    float z[8], minf[8];
    float m = -3.0e38f, cmax = -3.0e38f;
#pragma unroll
    for (int j = 0; j < 8; j++) {
        const int a = lane + 32 * j;
        z[j]    = Z[((size_t)(b * TA_ + a)) * TK_ + k];
        minf[j] = (1.f - mask_ans[b * TA_ + a]) * NEGC;
        m       = fmaxf(m, z[j] + minf[j]);
        cmax    = fmaxf(cmax, z[j] * invnk / na[b * TA_ + a] + minf[j]);
    }
    for (int o = 16; o > 0; o >>= 1) {
        m    = fmaxf(m, __shfl_xor(m, o, 32));
        cmax = fmaxf(cmax, __shfl_xor(cmax, o, 32));
    }
    float e[8], s = 0.f;
#pragma unroll
    for (int j = 0; j < 8; j++) {
        e[j] = __expf(z[j] + minf[j] - m);
        s += e[j];
    }
    for (int o = 16; o > 0; o >>= 1) s += __shfl_xor(s, o, 32);
    const float inv = 1.f / s;
#pragma unroll
    for (int j = 0; j < 8; j++) {
        const int a = lane + 32 * j;
        Pans[((size_t)(b * TA_ + a)) * TK_ + k] = e[j] * inv;
    }
    if (lane == 0) beta_ans[b * TK_ + k] = 1.f / (1.f + __expf(-5.f * cmax));
}

// ---------------------------------------------------------------------------
// K4/K5: out-tile = max over rows of ( (P @ Bsrc) * beta * maskRow ).
// P is indexed [b][kk][m] (so the GEMM A operand is P's transpose, staged by
// transposed LDS writes); Bsrc is [b][kk][E].  Full 256-row tiles so the
// token-axis max-pool finishes inside one workgroup -> no global atomics.
// A is single-term bf16 (probabilities), B is 2-term hi/lo split.
// ---------------------------------------------------------------------------
__global__ void __launch_bounds__(256)
pv_kernel(const float* __restrict__ P, const float* __restrict__ Bsrc,
          const float* __restrict__ maskB, const float* __restrict__ beta,
          const float* __restrict__ maskRow, float* __restrict__ fout) {
    __shared__ __align__(16) __bf16 As[256 * 32];
    __shared__ __align__(16) __bf16 Bh[64 * 32], Bl[64 * 32];
    __shared__ float red[8][2][64];

    const int b  = blockIdx.y;
    const int e0 = blockIdx.x * 64;
    const int tid = threadIdx.x;
    const int w = tid >> 5, lane = tid & 31;  // 8 waves x (32 rows x 64 cols)

    v8f acc[2][4];
#pragma unroll
    for (int i = 0; i < 2; i++)
#pragma unroll
        for (int j = 0; j < 4; j++) acc[i][j] = {};

    for (int k0 = 0; k0 < 256; k0 += 32) {
        // Stage A: P[b][k0+kk][m] -> As[m][kk] (transpose), 32x256.
#pragma unroll
        for (int q = 0; q < 8; q++) {
            const int idx  = tid + 256 * q;
            const int kk   = idx >> 6;
            const int mblk = (idx & 63) * 4;
            const float* pp = P + ((size_t)(b * 256 + k0 + kk)) * 256 + mblk;
            const float4 v = *(const float4*)pp;
            As[(mblk + 0) * 32 + kk] = (__bf16)v.x;
            As[(mblk + 1) * 32 + kk] = (__bf16)v.y;
            As[(mblk + 2) * 32 + kk] = (__bf16)v.z;
            As[(mblk + 3) * 32 + kk] = (__bf16)v.w;
            if (k0 + 32 < 256) prefetch_g(pp + 32 * 256);
        }
        // Stage B: Bsrc[b][k0+kk][e0+e] -> Bh/Bl[e][kk] (transpose), 32x64.
#pragma unroll
        for (int q = 0; q < 2; q++) {
            const int idx  = tid + 256 * q;
            const int kk   = idx >> 4;
            const int eblk = (idx & 15) * 4;
            const float mk = maskB[b * 256 + k0 + kk];
            const float* pb = Bsrc + ((size_t)(b * 256 + k0 + kk)) * E_ + e0 + eblk;
            const float4 v = *(const float4*)pb;
            float xs[4] = {v.x * mk, v.y * mk, v.z * mk, v.w * mk};
#pragma unroll
            for (int i = 0; i < 4; i++)
                split_bf16(xs[i], Bh[(eblk + i) * 32 + kk], Bl[(eblk + i) * 32 + kk]);
            if (k0 + 32 < 256) prefetch_g(pb + 32 * E_);
        }
        __syncthreads();

        v16bf af[2], bhf[4], blf[4];
#pragma unroll
        for (int i = 0; i < 2; i++) af[i] = ldfrag(As, w * 32 + 16 * i);
#pragma unroll
        for (int j = 0; j < 4; j++) {
            bhf[j] = ldfrag(Bh, 16 * j);
            blf[j] = ldfrag(Bl, 16 * j);
        }
#pragma unroll
        for (int i = 0; i < 2; i++)
#pragma unroll
            for (int j = 0; j < 4; j++) {
                acc[i][j] = wmma_bf16(af[i], bhf[j], acc[i][j]);
                acc[i][j] = wmma_bf16(af[i], blf[j], acc[i][j]);
            }
        __syncthreads();
    }

    // Epilogue: per-row scale, then column-max across 256 rows.
    float cm[4] = {-3.0e38f, -3.0e38f, -3.0e38f, -3.0e38f};
#pragma unroll
    for (int i = 0; i < 2; i++) {
        const int rowbase = w * 32 + 16 * i + (lane >> 4) * 8;
#pragma unroll
        for (int r = 0; r < 8; r++) {
            const int row = rowbase + r;
            const float s = maskRow[b * 256 + row] * beta[b * 256 + row];
#pragma unroll
            for (int j = 0; j < 4; j++) cm[j] = fmaxf(cm[j], acc[i][j][r] * s);
        }
    }
#pragma unroll
    for (int j = 0; j < 4; j++) red[w][lane >> 4][(lane & 15) + 16 * j] = cm[j];
    __syncthreads();
    if (tid < 64) {
        float m = -3.0e38f;
#pragma unroll
        for (int w2 = 0; w2 < 8; w2++)
            m = fmaxf(m, fmaxf(red[w2][0][tid], red[w2][1][tid]));
        fout[(size_t)b * (2 * E_) + e0 + tid] = m;
    }
}

// ---------------------------------------------------------------------------
// Launch
// ---------------------------------------------------------------------------
extern "C" void kernel_launch(void* const* d_in, const int* in_sizes, int n_in,
                              void* d_out, int out_size, void* d_ws, size_t ws_size,
                              hipStream_t stream) {
    (void)in_sizes; (void)n_in; (void)out_size; (void)ws_size;

    const float* ans      = (const float*)d_in[0];
    const float* mask_ans = (const float*)d_in[1];
    const float* key      = (const float*)d_in[2];
    const float* mask_key = (const float*)d_in[3];

    float* out = (float*)d_out;
    // Output layout (flat, return order):
    float* f        = out;                       // (B, 2E)   262144
    float* Z        = out + 262144;              // (B,TA,TK) 8388608
    float* Pans     = out + 8650752;             // (B,TA,TK) 8388608
    float* PkeyT    = out + 17039360;            // (B,TK,TA) 8388608
    float* beta_ans = out + 25427968;            // (B,1,TK)  32768
    float* beta_key = out + 25460736;            // (B,1,TA)  32768

    float* norm_ans = (float*)d_ws;              // B*TA
    float* norm_key = norm_ans + B_ * TA_;       // B*TK

    hipLaunchKernelGGL(norm_kernel, dim3((B_ * (TA_ + TK_)) / 8), dim3(256), 0, stream,
                       ans, mask_ans, key, mask_key, norm_ans, norm_key);

    hipLaunchKernelGGL(z_gemm_kernel, dim3(4, B_), dim3(256), 0, stream,
                       ans, key, mask_ans, mask_key, Z);

    hipLaunchKernelGGL(softmax_key_kernel, dim3(B_ * TA_ / 8), dim3(256), 0, stream,
                       Z, mask_key, norm_ans, norm_key, PkeyT, beta_key);

    hipLaunchKernelGGL(softmax_ans_kernel, dim3(B_ * TK_ / 8), dim3(256), 0, stream,
                       Z, mask_ans, norm_ans, norm_key, Pans, beta_ans);

    // V path: v = max_a( (P_key @ key) * beta_key * mask_ans ) -> f[:, E:2E]
    hipLaunchKernelGGL(pv_kernel, dim3(E_ / 64, B_), dim3(256), 0, stream,
                       PkeyT, key, mask_key, beta_key, mask_ans, f + E_);

    // U path: u = max_k( (P_ans^T @ ans) * beta_ans * mask_key ) -> f[:, 0:E]
    hipLaunchKernelGGL(pv_kernel, dim3(E_ / 64, B_), dim3(256), 0, stream,
                       Pans, ans, mask_ans, beta_ans, mask_key, f);
}
